// ReLU2SmearAssociativeLM_34368328303132
// MI455X (gfx1250) — matrix operand order repaired and checked by reading, not compiled, not run on hardware
//
#include <hip/hip_runtime.h>
#include <hip/hip_bf16.h>
#include <stdint.h>

// Problem constants (match reference)
#define B_  2
#define T_  2048
#define V_  32000
#define E_  512
#define H_  1024
#define MQ_ 128
#define SMEAR_C_ 24
#define NROW_ (B_ * T_)   // 4096 token rows

typedef __attribute__((ext_vector_type(16))) __bf16 v16bf;
typedef __attribute__((ext_vector_type(8)))  float  v8f;
typedef __hip_bfloat16 bf16_t;

// ---------------------------------------------------------------------------
// f32 -> bf16 conversion (weights, activations)
// ---------------------------------------------------------------------------
__global__ void f32_to_bf16_k(const float* __restrict__ in, bf16_t* __restrict__ out, int n) {
    int i = blockIdx.x * blockDim.x + threadIdx.x;
    if (i < n) out[i] = __float2bfloat16(in[i]);
}

// ---------------------------------------------------------------------------
// Transpose w_hh [3H,H] -> [H,3H] so the GRU scan reads lane-coalesced rows
// ---------------------------------------------------------------------------
__global__ void transpose_whh_k(const float* __restrict__ in, float* __restrict__ out) {
    int tid = blockIdx.x * blockDim.x + threadIdx.x;
    if (tid >= 3 * H_ * H_) return;
    int j = tid % (3 * H_);      // output column (gate row index in w_hh)
    int i = tid / (3 * H_);      // output row (h index)
    out[(size_t)i * (3 * H_) + j] = in[(size_t)j * H_ + i];
}

// ---------------------------------------------------------------------------
// Embedding gather + smear gate.  gate uses RAW current embedding (first 24
// channels) and adds gate * RAW previous embedding, matching the reference.
// ---------------------------------------------------------------------------
__global__ void embed_smear_k(const int*  __restrict__ ids,
                              const float* __restrict__ tab,
                              const float* __restrict__ smear_w,
                              const float* __restrict__ smear_lambda,
                              float* __restrict__ embF, bf16_t* __restrict__ embB) {
    int row = blockIdx.x;            // b*T + t
    int t   = row % T_;
    int id  = ids[row];
    const float* cur  = tab + (size_t)id * E_;
    const float* prev = nullptr;
    float gate = 0.0f;
    if (t > 0) {
        prev = tab + (size_t)ids[row - 1] * E_;
        float d = 0.0f;
        #pragma unroll
        for (int c = 0; c < SMEAR_C_; ++c) d = fmaf(cur[c], smear_w[c], d);
        gate = smear_lambda[0] * (1.0f / (1.0f + __expf(-d)));
    }
    for (int e = threadIdx.x; e < E_; e += blockDim.x) {
        float v = cur[e] + (t > 0 ? gate * prev[e] : 0.0f);
        embF[(size_t)row * E_ + e] = v;
        embB[(size_t)row * E_ + e] = __float2bfloat16(v);
    }
}

// ---------------------------------------------------------------------------
// Register-blocked bf16 WMMA GEMM:
//   out[M,N] = act( A[M,K] @ W[N,K]^T + bias[N] ),  act: 0=none, 1=relu^2
// One wave computes a 64x32 macro-tile = 4(M) x 2(N) WMMA sub-tiles.
// Per K=32 slab: 4 A-frags + 2 B-frags loaded (12x b128), 8 v_wmma issued
// -> ~22 FLOP/byte of fragment traffic (vs 8 for 1x1 tiling).
// Fragment layouts used here (contiguous k order, same for A and B):
//   A 16x32 bf16 : lane l -> row M=l&15, K-half (l>>4); dword vg holds K pair
//                  k = (l>>4)*16 + vg*2
//   B 32x16 bf16 : lane l -> col N=l&15, K group (l>>4)*16; dword vg holds
//                  K pair k = (l>>4)*16 + vg*2        (B[k][n] = W[n][k])
//   C/D 16x16 f32: vgpr r -> row (l>>4)*8 + r, col l&15
// ---------------------------------------------------------------------------
#define GM_ 4   // M sub-tiles per wave (64 rows)
#define GN_ 2   // N sub-tiles per wave (32 cols)

__global__ void __launch_bounds__(128)
gemm_wmma_bf16_k(const bf16_t* __restrict__ A, const bf16_t* __restrict__ W,
                 const float*  __restrict__ bias,
                 float* __restrict__ outF, bf16_t* __restrict__ outB,
                 int Mdim, int Ndim, int Kdim, int act) {
    const int lane   = threadIdx.x & 31;
    const int waveId = blockIdx.x * (blockDim.x >> 5) + (threadIdx.x >> 5);
    const int macroN = Ndim >> 5;                   // N / 32
    const int macroM = Mdim >> 6;                   // M / 64
    if (waveId >= macroM * macroN) return;          // uniform per wave
    const int tileM = (waveId / macroN) << 6;
    const int tileN = (waveId % macroN) << 5;

    const int mA    = lane & 15;
    const int khalf = lane >> 4;        // A: K-half select
    const int nW    = lane & 15;
    const int kgrp  = (lane >> 4) << 4; // B: K-group select (0 or 16)

    const uint32_t* Arow[GM_];
    #pragma unroll
    for (int i = 0; i < GM_; ++i)
        Arow[i] = (const uint32_t*)(A + (size_t)(tileM + (i << 4) + mA) * Kdim);
    const uint32_t* Wrow[GN_];
    #pragma unroll
    for (int j = 0; j < GN_; ++j)
        Wrow[j] = (const uint32_t*)(W + (size_t)(tileN + (j << 4) + nW) * Kdim);

    v8f acc[GM_][GN_] = {};
    for (int k0 = 0; k0 < Kdim; k0 += 32) {
        union Frag { v16bf v; uint32_t u[8]; };
        Frag a[GM_], b[GN_];
        #pragma unroll
        for (int j = 0; j < GN_; ++j)
            #pragma unroll
            for (int vg = 0; vg < 8; ++vg) {
                int kb = k0 + kgrp + (vg << 1);
                b[j].u[vg] = Wrow[j][kb >> 1];
            }
        #pragma unroll
        for (int i = 0; i < GM_; ++i)
            #pragma unroll
            for (int vg = 0; vg < 8; ++vg) {
                int ka = k0 + (khalf << 4) + (vg << 1);
                a[i].u[vg] = Arow[i][ka >> 1];
            }
        if (k0 + 32 < Kdim) {   // gfx1250 global_prefetch_b8 for the next K slab
            __builtin_prefetch((const void*)(Arow[0] + ((k0 + 32) >> 1)), 0, 0);
            __builtin_prefetch((const void*)(Wrow[0] + ((k0 + 32) >> 1)), 0, 0);
        }
        #pragma unroll
        for (int i = 0; i < GM_; ++i)
            #pragma unroll
            for (int j = 0; j < GN_; ++j)
                acc[i][j] = __builtin_amdgcn_wmma_f32_16x16x32_bf16(
                                false, a[i].v, false, b[j].v,
                                (short)0, acc[i][j], false, false);
    }

    #pragma unroll
    for (int j = 0; j < GN_; ++j) {
        const int nOut = tileN + (j << 4) + (lane & 15);
        const float bv = bias ? bias[nOut] : 0.0f;
        #pragma unroll
        for (int i = 0; i < GM_; ++i) {
            const int mBase = tileM + (i << 4) + ((lane >> 4) << 3);
            #pragma unroll
            for (int r = 0; r < 8; ++r) {
                float v = acc[i][j][r] + bv;
                if (act == 1) { v = fmaxf(v, 0.0f); v = v * v; }   // relu^2
                size_t idx = (size_t)(mBase + r) * Ndim + nOut;
                if (outF) outF[idx] = v;
                if (outB) outB[idx] = __float2bfloat16(v);
            }
        }
    }
}

// ---------------------------------------------------------------------------
// GRU scan (torch semantics).  One workgroup of 1024 threads (32 wave32s),
// h for both batches in LDS, weights pre-transposed to [H, 3H] so lane j's
// reads of column j / H+j / 2H+j are coalesced.  Thread j owns h[j] and the
// three gate dots for both batches (weight value reused across batches).
// ---------------------------------------------------------------------------
__global__ void __launch_bounds__(1024)
gru_scan_k(const float* __restrict__ xproj,   // [B*T, 3H]
           const float* __restrict__ whhT,    // [H, 3H]
           const float* __restrict__ b_hh,    // [3H]
           float* __restrict__ statesF,       // [B*T, H]
           bf16_t* __restrict__ statesB) {
    __shared__ float h0[H_];
    __shared__ float h1[H_];
    const int j = threadIdx.x;
    h0[j] = 0.0f; h1[j] = 0.0f;
    const float bhr = b_hh[j], bhz = b_hh[H_ + j], bhn = b_hh[2 * H_ + j];
    __syncthreads();

    for (int t = 0; t < T_; ++t) {
        float a0r = 0.f, a0z = 0.f, a0n = 0.f;
        float a1r = 0.f, a1z = 0.f, a1n = 0.f;
        #pragma unroll 4
        for (int i = 0; i < H_; ++i) {
            const float* wr = whhT + (size_t)i * (3 * H_);
            const float wRr = wr[j], wRz = wr[H_ + j], wRn = wr[2 * H_ + j];
            const float hv0 = h0[i], hv1 = h1[i];
            a0r = fmaf(hv0, wRr, a0r); a0z = fmaf(hv0, wRz, a0z); a0n = fmaf(hv0, wRn, a0n);
            a1r = fmaf(hv1, wRr, a1r); a1z = fmaf(hv1, wRz, a1z); a1n = fmaf(hv1, wRn, a1n);
        }
        const float* x0 = xproj + (size_t)t * (3 * H_);
        const float* x1 = xproj + (size_t)(T_ + t) * (3 * H_);
        float r0 = 1.0f / (1.0f + __expf(-(x0[j]          + a0r + bhr)));
        float z0 = 1.0f / (1.0f + __expf(-(x0[H_ + j]     + a0z + bhz)));
        float n0 = tanhf(x0[2 * H_ + j] + r0 * (a0n + bhn));
        float r1 = 1.0f / (1.0f + __expf(-(x1[j]          + a1r + bhr)));
        float z1 = 1.0f / (1.0f + __expf(-(x1[H_ + j]     + a1z + bhz)));
        float n1 = tanhf(x1[2 * H_ + j] + r1 * (a1n + bhn));
        float hn0 = (1.0f - z0) * n0 + z0 * h0[j];
        float hn1 = (1.0f - z1) * n1 + z1 * h1[j];
        __syncthreads();                     // all dots done before h update
        h0[j] = hn0; h1[j] = hn1;
        statesF[(size_t)t * H_ + j]         = hn0;
        statesF[(size_t)(T_ + t) * H_ + j]  = hn1;
        statesB[(size_t)t * H_ + j]         = __float2bfloat16(hn0);
        statesB[(size_t)(T_ + t) * H_ + j]  = __float2bfloat16(hn1);
        __syncthreads();                     // h visible before next step
    }
}

// ---------------------------------------------------------------------------
// g = sigmoid(states @ wg^T + bg)   (N=1: plain block reduction)
// ---------------------------------------------------------------------------
__global__ void __launch_bounds__(128)
gate_k(const float* __restrict__ states, const float* __restrict__ wg,
       const float* __restrict__ bg, float* __restrict__ g) {
    int row = blockIdx.x;
    __shared__ float red[128];
    float p = 0.0f;
    for (int i = threadIdx.x; i < H_; i += 128)
        p = fmaf(states[(size_t)row * H_ + i], wg[i], p);
    red[threadIdx.x] = p; __syncthreads();
    for (int off = 64; off > 0; off >>= 1) {
        if (threadIdx.x < off) red[threadIdx.x] += red[threadIdx.x + off];
        __syncthreads();
    }
    if (threadIdx.x == 0) g[row] = 1.0f / (1.0f + __expf(-(red[0] + bg[0])));
}

// ---------------------------------------------------------------------------
// Strictly-causal attention + vocab scatter.  softmax restricted to s<t
// equals the reference's masked-softmax-then-renormalize.  Scores staged in
// LDS; gated weights atomically added onto the logits row (ordered after the
// logits GEMM on the stream).
// ---------------------------------------------------------------------------
__global__ void __launch_bounds__(256)
attn_scatter_k(const float* __restrict__ q, const float* __restrict__ k,
               const float* __restrict__ g, const float* __restrict__ mscale,
               const int* __restrict__ ids, float* __restrict__ out) {
    int row = blockIdx.x;            // b*T + t
    int b = row / T_, t = row % T_;
    if (t == 0) return;              // no valid source tokens -> att row is 0
    __shared__ float qs[MQ_];
    __shared__ float sc[T_];
    __shared__ float red[256];
    int tid = threadIdx.x;
    for (int i = tid; i < MQ_; i += 256) qs[i] = q[(size_t)row * MQ_ + i];
    __syncthreads();

    float lmax = -3.4e38f;
    for (int s = tid; s < t; s += 256) {
        const float* kr = k + ((size_t)b * T_ + s) * MQ_;
        float d = 0.0f;
        #pragma unroll 8
        for (int jj = 0; jj < MQ_; ++jj) d = fmaf(qs[jj], kr[jj], d);
        d *= 0.08838834764831845f;   // 1/sqrt(128)
        sc[s] = d;
        lmax = fmaxf(lmax, d);
    }
    red[tid] = lmax; __syncthreads();
    for (int off = 128; off > 0; off >>= 1) {
        if (tid < off) red[tid] = fmaxf(red[tid], red[tid + off]);
        __syncthreads();
    }
    float mx = red[0]; __syncthreads();

    float lsum = 0.0f;
    for (int s = tid; s < t; s += 256) {
        float e = __expf(sc[s] - mx);
        sc[s] = e; lsum += e;
    }
    red[tid] = lsum; __syncthreads();
    for (int off = 128; off > 0; off >>= 1) {
        if (tid < off) red[tid] += red[tid + off];
        __syncthreads();
    }
    float S = fmaxf(red[0], 1e-6f);
    float coeff = g[row] * mscale[0] / S;

    float* orow = out + (size_t)row * V_;
    const int* idrow = ids + (size_t)b * T_;
    for (int s = tid; s < t; s += 256)
        atomicAdd(orow + idrow[s], sc[s] * coeff);
}

// ---------------------------------------------------------------------------
// Host-side orchestration
// ---------------------------------------------------------------------------
static inline size_t align256(size_t x) { return (x + 255) & ~(size_t)255; }

extern "C" void kernel_launch(void* const* d_in, const int* in_sizes, int n_in,
                              void* d_out, int out_size, void* d_ws, size_t ws_size,
                              hipStream_t stream) {
    (void)in_sizes; (void)n_in; (void)out_size;
    const int*   ids         = (const int*)  d_in[0];
    const float* emb_table   = (const float*)d_in[1];
    const float* w_ih        = (const float*)d_in[2];
    const float* w_hh        = (const float*)d_in[3];
    const float* b_ih        = (const float*)d_in[4];
    const float* b_hh        = (const float*)d_in[5];
    const float* wq          = (const float*)d_in[6];
    const float* bq          = (const float*)d_in[7];
    const float* wk          = (const float*)d_in[8];
    const float* bk          = (const float*)d_in[9];
    const float* wg          = (const float*)d_in[10];
    const float* bg          = (const float*)d_in[11];
    const float* w_fc        = (const float*)d_in[12];
    const float* b_fc        = (const float*)d_in[13];
    const float* w_proj      = (const float*)d_in[14];
    const float* b_proj      = (const float*)d_in[15];
    const float* output_bias = (const float*)d_in[16];
    const float* mscale      = (const float*)d_in[17];
    const float* smear_w     = (const float*)d_in[18];
    const float* smear_l     = (const float*)d_in[19];
    float* out = (float*)d_out;

    // workspace carve
    uint8_t* base = (uint8_t*)d_ws;
    size_t off = 0;
    auto carve = [&](size_t bytes) -> void* {
        void* r = base + off; off = align256(off + bytes); return r;
    };
    float*  embF    = (float*) carve((size_t)NROW_ * E_ * 4);
    bf16_t* embB    = (bf16_t*)carve((size_t)NROW_ * E_ * 2);
    float*  xproj   = (float*) carve((size_t)NROW_ * 3 * H_ * 4);
    float*  whhT    = (float*) carve((size_t)3 * H_ * H_ * 4);
    float*  statesF = (float*) carve((size_t)NROW_ * H_ * 4);
    bf16_t* statesB = (bf16_t*)carve((size_t)NROW_ * H_ * 2);
    bf16_t* headB   = (bf16_t*)carve((size_t)NROW_ * 4 * E_ * 2);
    bf16_t* featB   = (bf16_t*)carve((size_t)NROW_ * E_ * 2);
    float*  qF      = (float*) carve((size_t)NROW_ * MQ_ * 4);
    float*  kF      = (float*) carve((size_t)NROW_ * MQ_ * 4);
    float*  gF      = (float*) carve((size_t)NROW_ * 4);
    bf16_t* w_ihB   = (bf16_t*)carve((size_t)3 * H_ * E_ * 2);
    bf16_t* w_fcB   = (bf16_t*)carve((size_t)4 * E_ * H_ * 2);
    bf16_t* w_projB = (bf16_t*)carve((size_t)E_ * 4 * E_ * 2);
    bf16_t* tabB    = (bf16_t*)carve((size_t)V_ * E_ * 2);
    bf16_t* wqB     = (bf16_t*)carve((size_t)MQ_ * H_ * 2);
    bf16_t* wkB     = (bf16_t*)carve((size_t)MQ_ * H_ * 2);
    if (off > ws_size) return;   // workspace too small: do nothing (safe)

    auto conv = [&](const float* src, bf16_t* dst, int n) {
        f32_to_bf16_k<<<(n + 255) / 256, 256, 0, stream>>>(src, dst, n);
    };
    conv(w_ih,      w_ihB,   3 * H_ * E_);
    conv(w_fc,      w_fcB,   4 * E_ * H_);
    conv(w_proj,    w_projB, E_ * 4 * E_);
    conv(emb_table, tabB,    V_ * E_);
    conv(wq,        wqB,     MQ_ * H_);
    conv(wk,        wkB,     MQ_ * H_);

    {   // w_hh transpose for the scan
        int n = 3 * H_ * H_;
        transpose_whh_k<<<(n + 255) / 256, 256, 0, stream>>>(w_hh, whhT);
    }

    embed_smear_k<<<NROW_, 256, 0, stream>>>(ids, emb_table, smear_w, smear_l, embF, embB);

    auto gemm = [&](const bf16_t* A, const bf16_t* W, const float* bias,
                    float* oF, bf16_t* oB, int Md, int Nd, int Kd, int act) {
        int tiles = (Md >> 6) * (Nd >> 5);        // one wave per 64x32 macro-tile
        int blocks = (tiles + 3) / 4;             // 4 waves / 128-thread block
        gemm_wmma_bf16_k<<<blocks, 128, 0, stream>>>(A, W, bias, oF, oB, Md, Nd, Kd, act);
    };

    // x_proj = emb @ w_ih^T + b_ih
    gemm(embB, w_ihB, b_ih, xproj, nullptr, NROW_, 3 * H_, E_, 0);

    // sequential GRU scan
    gru_scan_k<<<1, 1024, 0, stream>>>(xproj, whhT, b_hh, statesF, statesB);

    // head = relu^2(states @ w_fc^T + b_fc)   (bf16 out only)
    gemm(statesB, w_fcB, b_fc, nullptr, headB, NROW_, 4 * E_, H_, 1);
    // feat = head @ w_proj^T + b_proj         (bf16 out only)
    gemm(headB, w_projB, b_proj, nullptr, featB, NROW_, E_, 4 * E_, 0);
    // logits = feat @ emb_table^T + output_bias  -> d_out (covers every element)
    gemm(featB, tabB, output_bias, out, nullptr, NROW_, V_, E_, 0);

    // q, k projections
    gemm(statesB, wqB, bq, qF, nullptr, NROW_, MQ_, H_, 0);
    gemm(statesB, wkB, bk, kF, nullptr, NROW_, MQ_, H_, 0);

    // scalar gate
    gate_k<<<NROW_, 128, 0, stream>>>(statesF, wg, bg, gF);

    // causal attention + scatter-add into logits
    attn_scatter_k<<<NROW_, 256, 0, stream>>>(qF, kF, gF, mscale, ids, out);
}
